// AtomLayer_51513837748549
// MI455X (gfx1250) — matrix operand order repaired and checked by reading, not compiled, not run on hardware
//
#include <hip/hip_runtime.h>
#include <hip/hip_bf16.h>

typedef __attribute__((ext_vector_type(16))) __bf16 v16bf;
typedef __attribute__((ext_vector_type(8)))  __bf16 v8bf;
typedef __attribute__((ext_vector_type(8)))  float  v8f;
typedef __attribute__((ext_vector_type(4)))  float  v4f;

#define D_ATOM 128
#define D_SPH  9
#define K1     265
#define K1PAD  288
#define H1     128
#define H2     64
#define H3     128

// LDS weight region sizes (bf16 elements), order: W1t G1t W2t G2t W3t G3t
#define SZ_W1T (H1 * K1PAD)   // 36864
#define SZ_W2T (H2 * H1)      // 8192
#define SZ_W3T (H3 * H2)      // 8192
#define W_ELEMS (2*SZ_W1T + 2*SZ_W2T + 2*SZ_W3T)   // 106496
#define W_BYTES (W_ELEMS * 2)                       // 212992

// per-wave LDS: h1[16][128] g1[16][128] h2[16][64] g2[16][64] ib[32] sb[16]
#define PW_BYTES (4096 + 4096 + 2048 + 2048 + 128 + 64)  // 12480
#define WAVES_PER_BLOCK 8
#define LDS_BYTES (W_BYTES + WAVES_PER_BLOCK * PW_BYTES) // 312832

__device__ __forceinline__ float siluf(float x) { return x / (1.0f + __expf(-x)); }
__device__ __forceinline__ float sigmf(float x) { return 1.0f / (1.0f + __expf(-x)); }

__device__ __forceinline__ v8f wmma_bf16(v16bf a, v16bf b, v8f c) {
    return __builtin_amdgcn_wmma_f32_16x16x32_bf16(false, a, false, b, (short)0, c, false, false);
}

// B fragment: weights stored transposed [N][kpad] bf16. Lane holds col n,
// lo: K = ks*32 + 8*kh + {0..7}; hi: K = ks*32 + 16 + 8*kh + {0..7} -> two ds_load_b128.
__device__ __forceinline__ v16bf loadB(const __bf16* base, int n, int kpad, int ks, int kh) {
    const __bf16* p = base + n * kpad + ks * 32 + kh * 8;
    v8bf lo = *(const v8bf*)p;
    v8bf hi = *(const v8bf*)(p + 16);
    v16bf r;
#pragma unroll
    for (int i = 0; i < 8; ++i) { r[i] = lo[i]; r[i + 8] = hi[i]; }
    return r;
}

// A fragment from LDS row-major [16][kpad] bf16. Lane m holds row m:
// lo: K = ks*32 + 8*kh + {0..7}; hi: K = ks*32 + 16 + 8*kh + {0..7}
__device__ __forceinline__ v16bf loadA_lds(const __bf16* base, int kpad, int ks, int m, int kh) {
    const __bf16* p = base + m * kpad + ks * 32 + 8 * kh;
    v8bf lo = *(const v8bf*)p;
    v8bf hi = *(const v8bf*)(p + 16);
    v16bf r;
#pragma unroll
    for (int i = 0; i < 8; ++i) { r[i] = lo[i]; r[i + 8] = hi[i]; }
    return r;
}

// A fragment straight from a global fp32 row (atom_attr is L2-resident), cvt->bf16
__device__ __forceinline__ v16bf loadA_g(const float* row, int k0) {
    v4f a0 = *(const v4f*)(row + k0);
    v4f a1 = *(const v4f*)(row + k0 + 4);
    v4f b0 = *(const v4f*)(row + k0 + 16);
    v4f b1 = *(const v4f*)(row + k0 + 20);
    v16bf r;
#pragma unroll
    for (int i = 0; i < 4; ++i) {
        r[i]      = (__bf16)a0[i];
        r[i + 4]  = (__bf16)a1[i];
        r[i + 8]  = (__bf16)b0[i];
        r[i + 12] = (__bf16)b1[i];
    }
    return r;
}

// ---------------- prep kernels ----------------

__global__ void __launch_bounds__(256) init_out_kernel(const float* __restrict__ atom_attr,
                                                       float* __restrict__ out, int n4) {
    int i = blockIdx.x * blockDim.x + threadIdx.x;
    if (i < n4) ((v4f*)out)[i] = ((const v4f*)atom_attr)[i];
}

__device__ __forceinline__ void pack_one(__bf16* dst, const float* src, int K, int N, int kpad, int idx) {
    int n = idx / kpad, k = idx % kpad;
    dst[idx] = (k < K) ? (__bf16)src[(size_t)k * N + n] : (__bf16)0.0f;
}

__global__ void __launch_bounds__(256) pack_weights_kernel(
    const float* __restrict__ W1, const float* __restrict__ G1,
    const float* __restrict__ W2, const float* __restrict__ G2,
    const float* __restrict__ W3, const float* __restrict__ G3,
    __bf16* __restrict__ wp) {
    int i = blockIdx.x * blockDim.x + threadIdx.x;
    if (i >= W_ELEMS) return;
    if (i < SZ_W1T)                 { pack_one(wp,                         W1, K1, H1, K1PAD, i); return; }
    i -= SZ_W1T;
    if (i < SZ_W1T)                 { pack_one(wp + SZ_W1T,               G1, K1, H1, K1PAD, i); return; }
    i -= SZ_W1T;
    if (i < SZ_W2T)                 { pack_one(wp + 2*SZ_W1T,             W2, H1, H2, H1, i); return; }
    i -= SZ_W2T;
    if (i < SZ_W2T)                 { pack_one(wp + 2*SZ_W1T + SZ_W2T,    G2, H1, H2, H1, i); return; }
    i -= SZ_W2T;
    if (i < SZ_W3T)                 { pack_one(wp + 2*SZ_W1T + 2*SZ_W2T,  W3, H2, H3, H2, i); return; }
    i -= SZ_W3T;
    pack_one(wp + 2*SZ_W1T + 2*SZ_W2T + SZ_W3T, G3, H2, H3, H2, i);
}

__global__ void __launch_bounds__(256) edge_scale_kernel(const float* __restrict__ ea,
                                                         const float* __restrict__ We,
                                                         const float* __restrict__ be,
                                                         float* __restrict__ sc, int E) {
    int e = blockIdx.x * blockDim.x + threadIdx.x;
    if (e >= E) return;
    const float* r = ea + (size_t)e * 100;
    float acc = be[0];
#pragma unroll
    for (int i = 0; i < 100; i += 4) {
        v4f v = *(const v4f*)(r + i);
        acc += v[0] * We[i] + v[1] * We[i + 1] + v[2] * We[i + 2] + v[3] * We[i + 3];
    }
    sc[e] = acc;
}

// ---------------- main fused edge-MLP kernel ----------------

__global__ void __launch_bounds__(256) edge_mlp_kernel(
    const float* __restrict__ atom_attr, const float* __restrict__ sph,
    const int* __restrict__ eidx, const float* __restrict__ scale,
    const __bf16* __restrict__ wp,
    const float* __restrict__ b1, const float* __restrict__ b2, const float* __restrict__ b3,
    const float* __restrict__ g1b, const float* __restrict__ g2b, const float* __restrict__ g3b,
    float* __restrict__ out, int nTiles, int E) {
    extern __shared__ char smem[];
    __bf16* lW1 = (__bf16*)smem;
    __bf16* lG1 = lW1 + SZ_W1T;
    __bf16* lW2 = lG1 + SZ_W1T;
    __bf16* lG2 = lW2 + SZ_W2T;
    __bf16* lW3 = lG2 + SZ_W2T;
    __bf16* lG3 = lW3 + SZ_W3T;

    const int wv   = threadIdx.x >> 5;
    const int lane = threadIdx.x & 31;
    const int m    = lane & 15;
    const int kh   = lane >> 4;

    char* pw   = smem + W_BYTES + wv * PW_BYTES;
    __bf16* h1 = (__bf16*)pw;
    __bf16* g1 = h1 + 16 * H1;
    __bf16* h2 = g1 + 16 * H1;
    __bf16* g2 = h2 + 16 * H2;
    int*    ib = (int*)(g2 + 16 * H2);
    float*  sb = (float*)(ib + 32);

    // stage all bf16 weights into LDS once per block (212992 B, b128 copies)
    {
        const uint4* s = (const uint4*)wp;
        uint4* d = (uint4*)smem;
        for (int i = threadIdx.x; i < W_BYTES / 16; i += blockDim.x) d[i] = s[i];
    }
    __syncthreads();

    // hoist biases (column depends only on lane + tile-col index)
    float b1v[8], g1v[8], b3v[8], g3v[8], b2v[4], g2v[4];
#pragma unroll
    for (int t = 0; t < 8; ++t) {
        int n = t * 16 + m;
        b1v[t] = b1[n]; g1v[t] = g1b[n];
        b3v[t] = b3[n]; g3v[t] = g3b[n];
    }
#pragma unroll
    for (int t = 0; t < 4; ++t) {
        int n = t * 16 + m;
        b2v[t] = b2[n]; g2v[t] = g2b[n];
    }

    const int tileStride = gridDim.x * WAVES_PER_BLOCK;

    for (int tile = blockIdx.x * WAVES_PER_BLOCK + wv; tile < nTiles; tile += tileStride) {
        int ebase = tile * 16;
        if (lane < 16) {
            int e = ebase + lane;
            int ec = e < E ? e : (E - 1);
            ib[lane]      = eidx[ec];
            ib[16 + lane] = eidx[(size_t)E + ec];
            sb[lane]      = (e < E) ? scale[e] : 0.0f;
        }
        // prefetch next tile's gather rows into cache (global_prefetch_b8)
        {
            int ntile = tile + tileStride;
            if (ntile < nTiles && lane < 16) {
                int ne = ntile * 16 + lane;
                int ns = eidx[ne];
                int nd = eidx[(size_t)E + ne];
                const float* pr0 = atom_attr + (size_t)ns * D_ATOM;
                const float* pr1 = atom_attr + (size_t)nd * D_ATOM;
                __builtin_prefetch(pr0, 0, 1);
                __builtin_prefetch(pr0 + 64, 0, 1);
                __builtin_prefetch(pr1, 0, 1);
                __builtin_prefetch(pr1 + 64, 0, 1);
            }
        }
        __builtin_amdgcn_wave_barrier();   // DS ops are in-order within a wave

        int e_m = ebase + m; if (e_m >= E) e_m = E - 1;
        const float* srcRow = atom_attr + (size_t)ib[m] * D_ATOM;
        const float* dstRow = atom_attr + (size_t)ib[16 + m] * D_ATOM;
        const float* sphRow = sph + (size_t)e_m * D_SPH;

        // ---- layer 1 (K=288): both branches share the gathered A fragment ----
        v8f accH[8], accG[8];
#pragma unroll
        for (int t = 0; t < 8; ++t) { accH[t] = {}; accG[t] = {}; }

#pragma unroll
        for (int ks = 0; ks < 9; ++ks) {
            v16bf a;
            if (ks < 8) {
                const float* row = (ks < 4) ? srcRow : dstRow;
                a = loadA_g(row, (ks & 3) * 32 + 8 * kh);
            } else {
                v16bf z;
#pragma unroll
                for (int i = 0; i < 8; ++i) {
                    int idx = 8 * kh + i;
                    z[i]     = (idx < D_SPH) ? (__bf16)sphRow[idx] : (__bf16)0.0f;
                    z[i + 8] = (__bf16)0.0f;   // K=272..287 padding
                }
                a = z;
            }
            // software-pipelined B fragments: load t+1 before wmma t
            v16bf bh = loadB(lW1, m, K1PAD, ks, kh);
            v16bf bg = loadB(lG1, m, K1PAD, ks, kh);
#pragma unroll
            for (int t = 0; t < 8; ++t) {
                v16bf bh_n, bg_n;
                if (t < 7) {
                    bh_n = loadB(lW1, (t + 1) * 16 + m, K1PAD, ks, kh);
                    bg_n = loadB(lG1, (t + 1) * 16 + m, K1PAD, ks, kh);
                } else {
                    bh_n = bh; bg_n = bg;
                }
                accH[t] = wmma_bf16(a, bh, accH[t]);
                accG[t] = wmma_bf16(a, bg, accG[t]);
                bh = bh_n; bg = bg_n;
            }
        }
        // bias + SiLU -> LDS (C layout: vgpr r -> row r + 8*kh, col = t*16+m)
#pragma unroll
        for (int t = 0; t < 8; ++t) {
            int n = t * 16 + m;
#pragma unroll
            for (int r = 0; r < 8; ++r) {
                int mo = r + 8 * kh;
                h1[mo * H1 + n] = (__bf16)siluf(accH[t][r] + b1v[t]);
                g1[mo * H1 + n] = (__bf16)siluf(accG[t][r] + g1v[t]);
            }
        }
        __builtin_amdgcn_wave_barrier();

        // ---- layer 2 (128 -> 64), pipelined over (ks, t) ----
        v8f aH2[4], aG2[4];
#pragma unroll
        for (int t = 0; t < 4; ++t) { aH2[t] = {}; aG2[t] = {}; }
#pragma unroll
        for (int ks = 0; ks < 4; ++ks) {
            v16bf ah = loadA_lds(h1, H1, ks, m, kh);
            v16bf ag = loadA_lds(g1, H1, ks, m, kh);
            v16bf bh = loadB(lW2, m, H1, ks, kh);
            v16bf bg = loadB(lG2, m, H1, ks, kh);
#pragma unroll
            for (int t = 0; t < 4; ++t) {
                v16bf bh_n, bg_n;
                if (t < 3) {
                    bh_n = loadB(lW2, (t + 1) * 16 + m, H1, ks, kh);
                    bg_n = loadB(lG2, (t + 1) * 16 + m, H1, ks, kh);
                } else {
                    bh_n = bh; bg_n = bg;
                }
                aH2[t] = wmma_bf16(ah, bh, aH2[t]);
                aG2[t] = wmma_bf16(ag, bg, aG2[t]);
                bh = bh_n; bg = bg_n;
            }
        }
#pragma unroll
        for (int t = 0; t < 4; ++t) {
            int n = t * 16 + m;
#pragma unroll
            for (int r = 0; r < 8; ++r) {
                int mo = r + 8 * kh;
                h2[mo * H2 + n] = (__bf16)siluf(aH2[t][r] + b2v[t]);
                g2[mo * H2 + n] = (__bf16)siluf(aG2[t][r] + g2v[t]);
            }
        }
        __builtin_amdgcn_wave_barrier();

        // ---- layer 3 (64 -> 128) fused with gate, scale, scatter-add ----
        // A fragments (4 total) are reused by all 8 column tiles: hoist them.
        v16bf a3h[2], a3g[2];
#pragma unroll
        for (int ks = 0; ks < 2; ++ks) {
            a3h[ks] = loadA_lds(h2, H2, ks, m, kh);
            a3g[ks] = loadA_lds(g2, H2, ks, m, kh);
        }
#pragma unroll
        for (int t = 0; t < 8; ++t) {
            v8f cH = {}, cG = {};
#pragma unroll
            for (int ks = 0; ks < 2; ++ks) {
                cH = wmma_bf16(a3h[ks], loadB(lW3, t * 16 + m, H2, ks, kh), cH);
                cG = wmma_bf16(a3g[ks], loadB(lG3, t * 16 + m, H2, ks, kh), cG);
            }
            int n = t * 16 + m;
#pragma unroll
            for (int r = 0; r < 8; ++r) {
                int mo = r + 8 * kh;
                float val = siluf(cH[r] + b3v[t]) * sigmf(cG[r] + g3v[t]) * sb[mo];
                int d = ib[16 + mo];
                unsafeAtomicAdd(out + (size_t)d * D_ATOM + n, val);
            }
        }
    }
}

// ---------------- launcher ----------------

extern "C" void kernel_launch(void* const* d_in, const int* in_sizes, int n_in,
                              void* d_out, int out_size, void* d_ws, size_t ws_size,
                              hipStream_t stream) {
    const float* atom_attr = (const float*)d_in[0];
    const float* edge_attr = (const float*)d_in[1];
    const float* sph       = (const float*)d_in[2];
    const int*   eidx      = (const int*)d_in[3];
    const float* W1 = (const float*)d_in[5];
    const float* b1 = (const float*)d_in[6];
    const float* W2 = (const float*)d_in[7];
    const float* b2 = (const float*)d_in[8];
    const float* W3 = (const float*)d_in[9];
    const float* b3 = (const float*)d_in[10];
    const float* G1 = (const float*)d_in[11];
    const float* g1 = (const float*)d_in[12];
    const float* G2 = (const float*)d_in[13];
    const float* g2 = (const float*)d_in[14];
    const float* G3 = (const float*)d_in[15];
    const float* g3 = (const float*)d_in[16];
    const float* We = (const float*)d_in[17];
    const float* be = (const float*)d_in[18];
    float* out = (float*)d_out;

    int natoms = in_sizes[0] / D_ATOM;
    int E      = in_sizes[3] / 2;
    int nTiles = (E + 15) / 16;

    float*  sc = (float*)d_ws;
    __bf16* wpk = (__bf16*)((char*)d_ws + (size_t)E * sizeof(float));

    // out = atom_attr
    int n4 = natoms * D_ATOM / 4;
    init_out_kernel<<<(n4 + 255) / 256, 256, 0, stream>>>(atom_attr, out, n4);
    // pack bf16 transposed/padded weights into workspace
    pack_weights_kernel<<<(W_ELEMS + 255) / 256, 256, 0, stream>>>(W1, G1, W2, G2, W3, G3, wpk);
    // per-edge scalar
    edge_scale_kernel<<<(E + 255) / 256, 256, 0, stream>>>(edge_attr, We, be, sc, E);

    // fused edge MLP + scatter
    static_assert(LDS_BYTES <= 320 * 1024, "LDS budget");
    hipFuncSetAttribute((const void*)edge_mlp_kernel,
                        hipFuncAttributeMaxDynamicSharedMemorySize, LDS_BYTES);
    int nblk = (nTiles + WAVES_PER_BLOCK - 1) / WAVES_PER_BLOCK;
    if (nblk > 2048) nblk = 2048;
    edge_mlp_kernel<<<nblk, 256, LDS_BYTES, stream>>>(
        atom_attr, sph, eidx, sc, wpk, b1, b2, b3, g1, g2, g3, out, nTiles, E);
}
